// AssociationBranch_46961172414954
// MI455X (gfx1250) — matrix operand order repaired and checked by reading, not compiled, not run on hardware
//
#include <hip/hip_runtime.h>

// ---------------------------------------------------------------------------
// MI455X (gfx1250) implementation of the two-branch GCN drug/cell model.
//
// Cost model: dense GEMMs total ~39 GFLOP (trivial for the WMMA pipe); the
// real cost is 4 edge-scatter passes of ~1.6 GB gather+atomic traffic each.
// xw / agg are 51.2 MB apiece -> both L2-resident (192 MB L2), so scatters run
// at L2 rather than HBM rates; HBM mostly streams the 12.8 MB edge-index
// arrays. Scatter maps one edge per wave32 (uniform src/dst/norm), float4 of
// channels per lane. GEMMs use V_WMMA_F32_16X16X32_F16 (f16 in, f32 acc) with
// the K loop fully unrolled (4/8/12 back-to-back WMMAs) and pure
// global_load_b128 operand feeds (NSEG templated so addrspace inference keeps
// every load on the global path instead of flat).
// ---------------------------------------------------------------------------

typedef __attribute__((ext_vector_type(16))) _Float16 v16h;
typedef __attribute__((ext_vector_type(8)))  float    v8f;

#define HIDC 128
#define NEG_SLOPE 0.2f
#define BN_EPS 1e-5f

// ---------------------------------------------------------------------------
// WMMA GEMM:  C[M x 128] = act( concat(A0..A_{NSEG-1})[M x NSEG*128] @ W + b )
// Each Ai is an M x 128 row-major f32 matrix; W is (NSEG*128) x 128 row-major.
// Block = 256 threads = 8 waves; wave w owns output cols [16w, 16w+16).
// ACT: 0 = none, 1 = leaky_relu(0.2), 2 = relu.  Requires M >= 16.
// ---------------------------------------------------------------------------
template <int NSEG, int ACT, bool BIAS>
__global__ __launch_bounds__(256) void gemm_f16wmma(
    const float* __restrict__ A0, const float* __restrict__ A1,
    const float* __restrict__ A2, int M,
    const float* __restrict__ W, const float* __restrict__ bias,
    float* __restrict__ C)
{
    const int lane = threadIdx.x & 31;
    const int wave = threadIdx.x >> 5;         // 0..7 -> output col tile
    const int m0   = blockIdx.x * 16;
    const int n0   = wave * 16;
    const int half = lane >> 4;                // 0: lanes 0-15, 1: lanes 16-31
    const int l15  = lane & 15;
    // Clamp OOB A rows (M % 16 == 0 for every call here anyway); OOB lanes
    // only contribute to OOB output rows, whose stores are guarded.
    const int rowA = min(m0 + l15, M - 1);
    const int colB = n0 + l15;

    v8f acc = {};
#pragma unroll
    for (int seg = 0; seg < NSEG; ++seg) {
        const float* __restrict__ A = (seg == 0) ? A0 : (seg == 1 ? A1 : A2);
        const float* __restrict__ arowbase = A + (size_t)rowA * HIDC + half * 16;
#pragma unroll
        for (int kk = 0; kk < HIDC; kk += 32) {
            // ---- A fragment (16x32 f16):
            //  lanes 0-15: K = kk+0..15; lanes 16-31: K = kk+16..31
            const float* arow = arowbase + kk;
            float4 f0 = *(const float4*)(arow + 0);
            float4 f1 = *(const float4*)(arow + 4);
            float4 f2 = *(const float4*)(arow + 8);
            float4 f3 = *(const float4*)(arow + 12);
            v16h a;
            a[0]=(_Float16)f0.x; a[1]=(_Float16)f0.y; a[2]=(_Float16)f0.z; a[3]=(_Float16)f0.w;
            a[4]=(_Float16)f1.x; a[5]=(_Float16)f1.y; a[6]=(_Float16)f1.z; a[7]=(_Float16)f1.w;
            a[8]=(_Float16)f2.x; a[9]=(_Float16)f2.y; a[10]=(_Float16)f2.z; a[11]=(_Float16)f2.w;
            a[12]=(_Float16)f3.x; a[13]=(_Float16)f3.y; a[14]=(_Float16)f3.z; a[15]=(_Float16)f3.w;

            // ---- B fragment (32x16 f16):
            //  lanes 0-15: K = kk+0..15; lanes 16-31: K = kk+16..31
            v16h b;
            const float* wp =
                W + (size_t)(seg * HIDC + kk + half * 16) * HIDC + colB;
#pragma unroll
            for (int j = 0; j < 16; ++j) b[j] = (_Float16)wp[(size_t)j * HIDC];

            acc = __builtin_amdgcn_wmma_f32_16x16x32_f16(
                /*neg_a=*/false, a, /*neg_b=*/false, b,
                /*c_mod=*/(short)0, acc, /*reuse_a=*/false, /*reuse_b=*/false);
        }
    }
    // ---- epilogue: C/D layout VGPR r -> M = r (lanes0-15) or 8+r (lanes16-31)
#pragma unroll
    for (int r = 0; r < 8; ++r) {
        int row = m0 + half * 8 + r;
        if (row < M) {
            float v = acc[r];
            if (BIAS) v += bias[colB];
            if (ACT == 1)      v = (v > 0.f) ? v : NEG_SLOPE * v;
            else if (ACT == 2) v = fmaxf(v, 0.f);
            C[(size_t)row * HIDC + colB] = v;
        }
    }
}

// ---------------------------------------------------------------------------
// Degree accumulation: deg[dst[e]] += 1
// ---------------------------------------------------------------------------
__global__ __launch_bounds__(256) void deg_kernel(
    const int* __restrict__ dst, int E, float* __restrict__ deg)
{
    int i = blockIdx.x * blockDim.x + threadIdx.x;
    if (i < E) atomicAdd(&deg[dst[i]], 1.0f);
}

// deg -> d^{-1/2} in place
__global__ __launch_bounds__(256) void dis_kernel(float* __restrict__ deg, int n)
{
    int i = blockIdx.x * blockDim.x + threadIdx.x;
    if (i < n) {
        float d = deg[i];
        deg[i] = (d > 0.f) ? rsqrtf(d) : 0.f;
    }
}

// ---------------------------------------------------------------------------
// Edge scatter: agg[dst] += (dis[src]*dis[dst]) * xw[src]
// One edge per wave32; each lane handles 4 channels (float4 gather).
// xw and agg are 51 MB each -> L2-resident on MI455X (192 MB L2).
// ---------------------------------------------------------------------------
__global__ __launch_bounds__(256) void scatter_kernel(
    const int* __restrict__ src, const int* __restrict__ dst, int E,
    const float* __restrict__ dis, const float* __restrict__ xw,
    float* __restrict__ agg)
{
    int e    = (int)((blockIdx.x * (unsigned)blockDim.x + threadIdx.x) >> 5);
    int lane = threadIdx.x & 31;
    if (e >= E) return;
    int s = src[e];                 // wave-uniform broadcast loads
    int d = dst[e];
    float nrm = dis[s] * dis[d];
    float4 v = ((const float4*)(xw + (size_t)s * HIDC))[lane];
    float* out = agg + (size_t)d * HIDC + lane * 4;
    atomicAdd(out + 0, nrm * v.x);
    atomicAdd(out + 1, nrm * v.y);
    atomicAdd(out + 2, nrm * v.z);
    atomicAdd(out + 3, nrm * v.w);
}

// out = leaky_relu(agg + bias)  (elementwise over M x 128)
__global__ __launch_bounds__(256) void bias_act_kernel(
    const float* __restrict__ agg, const float* __restrict__ b,
    float* __restrict__ out, size_t total)
{
    size_t i = blockIdx.x * (size_t)blockDim.x + threadIdx.x;
    if (i >= total) return;
    float v = agg[i] + b[i & (HIDC - 1)];
    out[i] = (v > 0.f) ? v : NEG_SLOPE * v;
}

// Column partial sums for BatchNorm (biased variance, training mode)
__global__ __launch_bounds__(128) void bn_stats_kernel(
    const float* __restrict__ x, int M,
    float* __restrict__ sum, float* __restrict__ sq)
{
    int col = threadIdx.x;          // 128 columns
    float s = 0.f, q = 0.f;
    for (int r = blockIdx.x; r < M; r += gridDim.x) {
        float v = x[(size_t)r * HIDC + col];
        s += v; q += v * v;
    }
    atomicAdd(&sum[col], s);
    atomicAdd(&sq[col],  q);
}

// In-place BN apply: y = (x - mu) * rsqrt(var + eps) * g + be
__global__ __launch_bounds__(256) void bn_apply_kernel(
    float* __restrict__ x, const float* __restrict__ sum,
    const float* __restrict__ sq, const float* __restrict__ g,
    const float* __restrict__ be, int M)
{
    size_t total = (size_t)M * HIDC;
    size_t i = blockIdx.x * (size_t)blockDim.x + threadIdx.x;
    if (i >= total) return;
    int col = (int)(i & (HIDC - 1));
    float inv = 1.0f / (float)M;
    float mu  = sum[col] * inv;
    float var = sq[col] * inv - mu * mu;
    float sc  = g[col] * rsqrtf(var + BN_EPS);
    x[i] = (x[i] - mu) * sc + be[col];
}

// Zero fill (graph-capture-safe alternative to memset)
__global__ __launch_bounds__(256) void fill0_kernel(float* __restrict__ p, size_t n)
{
    size_t i = blockIdx.x * (size_t)blockDim.x + threadIdx.x;
    if (i < n) p[i] = 0.f;
}

// ---------------------------------------------------------------------------
extern "C" void kernel_launch(void* const* d_in, const int* in_sizes, int n_in,
                              void* d_out, int out_size, void* d_ws, size_t ws_size,
                              hipStream_t stream)
{
    (void)n_in; (void)out_size; (void)ws_size;

    const float* feature = (const float*)d_in[0];
    const int*   sen     = (const int*)d_in[1];   // [2, E]: row0=src, row1=dst
    const int*   res     = (const int*)d_in[2];
    const int Nn = in_sizes[0] / HIDC;            // 100000
    const int Ee = in_sizes[1] / 2;               // 1600000

    const float *W_sen1 = (const float*)d_in[3],  *b_sen1 = (const float*)d_in[4];
    const float *W_sen2 = (const float*)d_in[5],  *b_sen2 = (const float*)d_in[6];
    const float *W_senfc= (const float*)d_in[7],  *b_senfc= (const float*)d_in[8];
    const float *g_sen  = (const float*)d_in[9],  *be_sen = (const float*)d_in[10];
    const float *W_res1 = (const float*)d_in[11], *b_res1 = (const float*)d_in[12];
    const float *W_res2 = (const float*)d_in[13], *b_res2 = (const float*)d_in[14];
    const float *W_resfc= (const float*)d_in[15], *b_resfc= (const float*)d_in[16];
    const float *g_res  = (const float*)d_in[17], *be_res = (const float*)d_in[18];
    const float *W_cell = (const float*)d_in[19], *b_cell = (const float*)d_in[20];
    const float *W_drug = (const float*)d_in[21], *b_drug = (const float*)d_in[22];

    const size_t NH = (size_t)Nn * HIDC;
    float* ws       = (float*)d_ws;
    float* dis      = ws;                 // N        (deg -> d^-1/2 in place)
    float* xw       = dis + Nn;           // N x 128
    float* agg      = xw  + NH;           // N x 128
    float* x1       = agg + NH;           // N x 128
    float* x2       = x1  + NH;           // N x 128
    float* xall_sen = x2  + NH;           // N x 128  (persists across branches)
    float* xall_res = xall_sen + NH;      // N x 128
    float* bnsum    = xall_res + NH;      // 128 sums + 128 sumsq (contiguous)

    const int gTiles = (Nn + 15) / 16;
    const int gNH    = (int)((NH + 255) / 256);
    const int gE     = (Ee + 255) / 256;
    const int gN     = (Nn + 255) / 256;
    const int gScat  = (Ee + 7) / 8;      // 8 edges (waves) per 256-thread block

    auto run_branch = [&](const int* edges,
                          const float* W1, const float* b1,
                          const float* W2, const float* b2,
                          const float* Wfc, const float* bfc,
                          const float* g, const float* be, float* xall)
    {
        const int* src = edges;
        const int* dst = edges + Ee;

        // symmetric edge normalization: dis = d^-1/2 at each node
        fill0_kernel<<<gN, 256, 0, stream>>>(dis, (size_t)Nn);
        deg_kernel<<<gE, 256, 0, stream>>>(dst, Ee, dis);
        dis_kernel<<<gN, 256, 0, stream>>>(dis, Nn);

        // GCN layer 1: xw = feature @ W1 ; agg = scatter ; x1 = lrelu(agg+b1)
        gemm_f16wmma<1, 0, false><<<gTiles, 256, 0, stream>>>(
            feature, nullptr, nullptr, Nn, W1, nullptr, xw);
        fill0_kernel<<<gNH, 256, 0, stream>>>(agg, NH);
        scatter_kernel<<<gScat, 256, 0, stream>>>(src, dst, Ee, dis, xw, agg);
        bias_act_kernel<<<gNH, 256, 0, stream>>>(agg, b1, x1, NH);

        // GCN layer 2
        gemm_f16wmma<1, 0, false><<<gTiles, 256, 0, stream>>>(
            x1, nullptr, nullptr, Nn, W2, nullptr, xw);
        fill0_kernel<<<gNH, 256, 0, stream>>>(agg, NH);
        scatter_kernel<<<gScat, 256, 0, stream>>>(src, dst, Ee, dis, xw, agg);
        bias_act_kernel<<<gNH, 256, 0, stream>>>(agg, b2, x2, NH);

        // FC over concat [feature, x1, x2] (K = 384), ReLU
        gemm_f16wmma<3, 2, true><<<gTiles, 256, 0, stream>>>(
            feature, x1, x2, Nn, Wfc, bfc, xall);

        // BatchNorm (training mode, biased variance), in place
        fill0_kernel<<<1, 256, 0, stream>>>(bnsum, (size_t)(2 * HIDC));
        bn_stats_kernel<<<512, 128, 0, stream>>>(xall, Nn, bnsum, bnsum + HIDC);
        bn_apply_kernel<<<gNH, 256, 0, stream>>>(xall, bnsum, bnsum + HIDC,
                                                 g, be, Nn);
    };

    run_branch(sen, W_sen1, b_sen1, W_sen2, b_sen2, W_senfc, b_senfc,
               g_sen, be_sen, xall_sen);
    run_branch(res, W_res1, b_res1, W_res2, b_res2, W_resfc, b_resfc,
               g_res, be_res, xall_res);

    // Heads: rows [0,40000) -> cell head, rows [40000,N) -> drug head.
    // K = 256 = concat(x_sen, x_res) segments.
    const int NUM_CELL = 40000;
    const int num_drug = Nn - NUM_CELL;
    float* out = (float*)d_out;

    gemm_f16wmma<2, 2, true><<<(NUM_CELL + 15) / 16, 256, 0, stream>>>(
        xall_sen, xall_res, nullptr, NUM_CELL, W_cell, b_cell, out);

    gemm_f16wmma<2, 2, true><<<(num_drug + 15) / 16, 256, 0, stream>>>(
        xall_sen + (size_t)NUM_CELL * HIDC,
        xall_res + (size_t)NUM_CELL * HIDC, nullptr,
        num_drug, W_drug, b_drug, out + (size_t)NUM_CELL * HIDC);
}
